// GHConv_29008209117473
// MI455X (gfx1250) — matrix operand order repaired and checked by reading, not compiled, not run on hardware
//
#include <hip/hip_runtime.h>
#include <hip/hip_bf16.h>

// ---------- types for WMMA ----------
typedef __attribute__((ext_vector_type(16))) __bf16 v16bf;
typedef __attribute__((ext_vector_type(8)))  float  v8f;

union BF16x16 {
    v16bf v;
    unsigned short u[16];
    uint4 q[2];
};

// f32 -> bf16 with round-to-nearest-even
static __device__ __forceinline__ unsigned short f2bf(float f) {
    unsigned int x = __float_as_uint(f);
    unsigned int r = x + 0x7FFFu + ((x >> 16) & 1u);
    return (unsigned short)(r >> 16);
}

static __device__ __forceinline__ v8f wmma_bf16(const BF16x16& a, const BF16x16& b, v8f c) {
    return __builtin_amdgcn_wmma_f32_16x16x32_bf16(
        /*neg_a=*/false, a.v, /*neg_b=*/false, b.v,
        /*c_mod=*/(short)0, c, /*reuse_a=*/false, /*reuse_b=*/false);
}

#define BATCH 4
#define NNODE 4096
#define DDIM  256

// ---------------------------------------------------------------------------
// Kernel 1: convert the three DxD weight matrices to bf16, transposed [e][k]
// so that WMMA B-fragments are contiguous 32-byte per-lane loads.
// ---------------------------------------------------------------------------
__global__ __launch_bounds__(256) void convert_weights_kernel(
    const float* __restrict__ W_t, const float* __restrict__ W_h,
    const float* __restrict__ Th,
    unsigned short* __restrict__ wtT, unsigned short* __restrict__ whT,
    unsigned short* __restrict__ thT)
{
    int i = blockIdx.x * blockDim.x + threadIdx.x;   // 0 .. 65535
    if (i >= DDIM * DDIM) return;
    int k = i >> 8;        // source row (K dim)
    int e = i & 255;       // source col (N dim)
    int o = e * DDIM + k;  // transposed
    wtT[o] = f2bf(W_t[i]);
    whT[o] = f2bf(W_h[i]);
    thT[o] = f2bf(Th[i]);
}

// ---------------------------------------------------------------------------
// Kernel 2: degree + norm = rsqrt(sum_m adj[b,n,m] + 1e-6).
// One wave32 per adjacency row; float4 loads + shuffle reduction.
// ---------------------------------------------------------------------------
__global__ __launch_bounds__(256) void degree_norm_kernel(
    const float* __restrict__ adj, float* __restrict__ norm)
{
    int row  = blockIdx.x * 8 + (threadIdx.x >> 5);  // B*N rows total
    int lane = threadIdx.x & 31;
    const float4* p = (const float4*)(adj + (size_t)row * NNODE);
    float s = 0.f;
    for (int j = lane; j < NNODE / 4; j += 32) {
        float4 v = p[j];
        s += v.x + v.y + v.z + v.w;
    }
    for (int off = 16; off > 0; off >>= 1) s += __shfl_down(s, off, 32);
    if (lane == 0) norm[row] = rsqrtf(s + 1e-6f);
}

// ---------------------------------------------------------------------------
// Kernel 3: the three x-projections via WMMA (K = 256).
// WG = 16 rows of x (one batch), 8 waves, wave w owns column tiles 2w, 2w+1.
//   theta path -> tT  = bf16 transposed [b][e][m] of (x@theta)*norm[m]
//   W_h  path -> f_het (f32)
//   W_t  path -> gate = sigmoid(x@W_t + b_t)  (f32)
// ---------------------------------------------------------------------------
__global__ __launch_bounds__(256) void proj_kernel(
    const float* __restrict__ x, const float* __restrict__ b_t,
    const float* __restrict__ norm,
    const unsigned short* __restrict__ thT, const unsigned short* __restrict__ whT,
    const unsigned short* __restrict__ wtT,
    unsigned short* __restrict__ tT, float* __restrict__ f_het,
    float* __restrict__ gate)
{
    __shared__ alignas(16) unsigned short xs[16 * DDIM];  // bf16 x tile [16][256]

    int blk = blockIdx.x;              // 0..1023
    int b   = blk >> 8;                // 256 row-blocks per batch
    int rb  = blk & 255;
    size_t rbase = (size_t)b * NNODE + rb * 16;   // flat row index base

    for (int i = threadIdx.x; i < 16 * DDIM; i += 256) {
        int r = i >> 8, c = i & 255;
        xs[i] = f2bf(x[(rbase + r) * DDIM + c]);
    }
    __syncthreads();

    int wave = threadIdx.x >> 5;
    int lane = threadIdx.x & 31;
    int m    = lane & 15;
    int half = lane >> 4;

    for (int wsel = 0; wsel < 3; ++wsel) {
        const unsigned short* Wm = (wsel == 0) ? thT : ((wsel == 1) ? whT : wtT);
        v8f acc0 = {}, acc1 = {};
        int c0 = wave * 32;        // column tile 0 base
        int c1 = c0 + 16;          // column tile 1 base
        for (int ks = 0; ks < DDIM; ks += 32) {
            // A fragment from LDS:
            //   e=0..15 -> K = ks + 16*half + e
            BF16x16 A;
            A.q[0] = *(const uint4*)&xs[m * DDIM + ks + half * 16];
            A.q[1] = *(const uint4*)&xs[m * DDIM + ks + half * 16 + 8];
            // B fragments (transposed weight, [col][k]): lane=col, e<->K,
            // K half-split at 16: K = ks + 16*half + e  (32 contiguous bytes)
            BF16x16 B0, B1;
            const uint4* pb0 = (const uint4*)&Wm[(size_t)(c0 + m) * DDIM + ks + half * 16];
            const uint4* pb1 = (const uint4*)&Wm[(size_t)(c1 + m) * DDIM + ks + half * 16];
            B0.q[0] = pb0[0]; B0.q[1] = pb0[1];
            B1.q[0] = pb1[0]; B1.q[1] = pb1[1];
            acc0 = wmma_bf16(A, B0, acc0);
            acc1 = wmma_bf16(A, B1, acc1);
        }
        // Epilogue. C/D layout: element r -> row (8*half + r), lane&15 -> col.
        if (wsel == 0) {
            // tT[b][col][m] = bf16( acc * norm[row] ), rows contiguous per lane
            unsigned short tmp0[8], tmp1[8];
            #pragma unroll
            for (int r = 0; r < 8; ++r) {
                float nr = norm[rbase + 8 * half + r];
                tmp0[r] = f2bf(acc0[r] * nr);
                tmp1[r] = f2bf(acc1[r] * nr);
            }
            size_t row0 = (size_t)rb * 16 + 8 * half;
            *(uint4*)&tT[((size_t)b * DDIM + c0 + m) * NNODE + row0] = *(const uint4*)tmp0;
            *(uint4*)&tT[((size_t)b * DDIM + c1 + m) * NNODE + row0] = *(const uint4*)tmp1;
        } else if (wsel == 1) {
            #pragma unroll
            for (int r = 0; r < 8; ++r) {
                size_t n = rbase + 8 * half + r;
                f_het[n * DDIM + c0 + m] = acc0[r];
                f_het[n * DDIM + c1 + m] = acc1[r];
            }
        } else {
            float bt0 = b_t[c0 + m];
            float bt1 = b_t[c1 + m];
            #pragma unroll
            for (int r = 0; r < 8; ++r) {
                size_t n = rbase + 8 * half + r;
                gate[n * DDIM + c0 + m] = 1.f / (1.f + __expf(-(acc0[r] + bt0)));
                gate[n * DDIM + c1 + m] = 1.f / (1.f + __expf(-(acc1[r] + bt1)));
            }
        }
    }
}

// ---------------------------------------------------------------------------
// Kernel 4: fused SpMM (adj @ t) * norm, gate blend, ELU.
// WG = 16 output rows x 256 cols of one batch; K loop over 4096.
// adj read exactly once from HBM (f32 -> bf16 in registers).
// ---------------------------------------------------------------------------
__global__ __launch_bounds__(256) void spmm_fuse_kernel(
    const float* __restrict__ adj, const unsigned short* __restrict__ tT,
    const float* __restrict__ norm, const float* __restrict__ f_het,
    const float* __restrict__ gate, float* __restrict__ out)
{
    int blk = blockIdx.x;              // 0..1023
    int b   = blk >> 8;
    int rb  = blk & 255;

    int wave = threadIdx.x >> 5;
    int lane = threadIdx.x & 31;
    int m    = lane & 15;
    int half = lane >> 4;

    size_t adj_row = ((size_t)b * NNODE + (size_t)rb * 16 + m) * NNODE;
    const unsigned short* tb = tT + (size_t)b * DDIM * NNODE;
    int c0 = wave * 32;
    int c1 = c0 + 16;

    v8f acc0 = {}, acc1 = {};
    for (int ks = 0; ks < NNODE; ks += 32) {
        // A fragment: adj[row rb*16+m][ks + 16*half + 0..15]
        const float* ap = adj + adj_row + ks;
        float4 a0 = *(const float4*)(ap + half * 16);
        float4 a1 = *(const float4*)(ap + half * 16 + 4);
        float4 a2 = *(const float4*)(ap + half * 16 + 8);
        float4 a3 = *(const float4*)(ap + half * 16 + 12);
        BF16x16 A;
        A.u[0]  = f2bf(a0.x); A.u[1]  = f2bf(a0.y); A.u[2]  = f2bf(a0.z); A.u[3]  = f2bf(a0.w);
        A.u[4]  = f2bf(a1.x); A.u[5]  = f2bf(a1.y); A.u[6]  = f2bf(a1.z); A.u[7]  = f2bf(a1.w);
        A.u[8]  = f2bf(a2.x); A.u[9]  = f2bf(a2.y); A.u[10] = f2bf(a2.z); A.u[11] = f2bf(a2.w);
        A.u[12] = f2bf(a3.x); A.u[13] = f2bf(a3.y); A.u[14] = f2bf(a3.z); A.u[15] = f2bf(a3.w);
        // B fragments from tT (lane = col, 16 contiguous K per lane-half)
        BF16x16 B0, B1;
        const uint4* pb0 = (const uint4*)&tb[(size_t)(c0 + m) * NNODE + ks + half * 16];
        const uint4* pb1 = (const uint4*)&tb[(size_t)(c1 + m) * NNODE + ks + half * 16];
        B0.q[0] = pb0[0]; B0.q[1] = pb0[1];
        B1.q[0] = pb1[0]; B1.q[1] = pb1[1];
        acc0 = wmma_bf16(A, B0, acc0);
        acc1 = wmma_bf16(A, B1, acc1);
    }

    // Epilogue: out = elu( gate * fhom * norm[n] + (1-gate) * f_het )
    #pragma unroll
    for (int r = 0; r < 8; ++r) {
        size_t n  = (size_t)b * NNODE + (size_t)rb * 16 + 8 * half + r;
        float  nr = norm[n];
        {
            size_t o = n * DDIM + c0 + m;
            float g = gate[o];
            float v = g * (acc0[r] * nr) + (1.f - g) * f_het[o];
            out[o] = v > 0.f ? v : expm1f(v);
        }
        {
            size_t o = n * DDIM + c1 + m;
            float g = gate[o];
            float v = g * (acc1[r] * nr) + (1.f - g) * f_het[o];
            out[o] = v > 0.f ? v : expm1f(v);
        }
    }
}

// ---------------------------------------------------------------------------
// Launch
// ---------------------------------------------------------------------------
extern "C" void kernel_launch(void* const* d_in, const int* in_sizes, int n_in,
                              void* d_out, int out_size, void* d_ws, size_t ws_size,
                              hipStream_t stream)
{
    const float* x   = (const float*)d_in[0];
    const float* adj = (const float*)d_in[1];
    const float* W_t = (const float*)d_in[2];
    const float* b_t = (const float*)d_in[3];
    const float* W_h = (const float*)d_in[4];
    const float* th  = (const float*)d_in[5];
    float* out = (float*)d_out;

    char* ws = (char*)d_ws;
    // workspace layout (bytes)
    float*          norm  = (float*)(ws + 0);                  //    65,536
    unsigned short* tT    = (unsigned short*)(ws + 65536);     // 8,388,608
    float*          f_het = (float*)(ws + 8454144);            // 16,777,216
    float*          gate  = (float*)(ws + 25231360);           // 16,777,216
    unsigned short* thT   = (unsigned short*)(ws + 42008576);  //   131,072
    unsigned short* whT   = thT + DDIM * DDIM;                 //   131,072
    unsigned short* wtT   = whT + DDIM * DDIM;                 //   131,072

    convert_weights_kernel<<<256, 256, 0, stream>>>(W_t, W_h, th, wtT, whT, thT);
    degree_norm_kernel<<<(BATCH * NNODE) / 8, 256, 0, stream>>>(adj, norm);
    proj_kernel<<<BATCH * (NNODE / 16), 256, 0, stream>>>(
        x, b_t, norm, thT, whT, wtT, tT, f_het, gate);
    spmm_fuse_kernel<<<BATCH * (NNODE / 16), 256, 0, stream>>>(
        adj, tT, norm, f_het, gate, out);
}
